// VectorQuantizer_30245159699092
// MI455X (gfx1250) — matrix operand order repaired and checked by reading, not compiled, not run on hardware
//
#include <hip/hip_runtime.h>
#include <hip/hip_bf16.h>
#include <stdint.h>

typedef __attribute__((ext_vector_type(16))) __bf16 v16bf;
typedef __attribute__((ext_vector_type(8)))  float  v8f;

#define DIM     256
#define NZ      16384          // 16*1024 z rows
#define NK      8192           // codebook rows
#define KTILES  8              // 256 / 32
#define NTILES  (NK / 16)      // 512
#define ZQ_ELEMS ((size_t)NZ * DIM)   // 4194304

__device__ __forceinline__ unsigned short f32_to_bf16_rne(float f) {
    union { float f; uint32_t u; } v; v.f = f;
    uint32_t u = v.u;
    u += 0x7FFFu + ((u >> 16) & 1u);
    return (unsigned short)(u >> 16);
}

// ---------------------------------------------------------------------------
// Normalize rows of a [nrows x 256] f32 matrix, emit bf16. One wave per row.
// ---------------------------------------------------------------------------
__global__ __launch_bounds__(128) void vq_normalize_bf16(
    const float* __restrict__ src, unsigned short* __restrict__ dst, int nrows) {
    const int wave = threadIdx.x >> 5;
    const int lane = threadIdx.x & 31;
    const int row  = blockIdx.x * 4 + wave;
    if (row >= nrows) return;

    const float4* r4 = (const float4*)(src + (size_t)row * DIM);
    float4 a = r4[lane * 2 + 0];
    float4 b = r4[lane * 2 + 1];

    float s = a.x*a.x + a.y*a.y + a.z*a.z + a.w*a.w
            + b.x*b.x + b.y*b.y + b.z*b.z + b.w*b.w;
    #pragma unroll
    for (int m = 16; m >= 1; m >>= 1) s += __shfl_xor(s, m, 32);

    const float scale = 1.0f / fmaxf(sqrtf(s), 1e-12f);

    unsigned short h[8];
    h[0] = f32_to_bf16_rne(a.x * scale); h[1] = f32_to_bf16_rne(a.y * scale);
    h[2] = f32_to_bf16_rne(a.z * scale); h[3] = f32_to_bf16_rne(a.w * scale);
    h[4] = f32_to_bf16_rne(b.x * scale); h[5] = f32_to_bf16_rne(b.y * scale);
    h[6] = f32_to_bf16_rne(b.z * scale); h[7] = f32_to_bf16_rne(b.w * scale);

    uint4 packed;
    packed.x = (uint32_t)h[0] | ((uint32_t)h[1] << 16);
    packed.y = (uint32_t)h[2] | ((uint32_t)h[3] << 16);
    packed.z = (uint32_t)h[4] | ((uint32_t)h[5] << 16);
    packed.w = (uint32_t)h[6] | ((uint32_t)h[7] << 16);
    ((uint4*)(dst + (size_t)row * DIM))[lane] = packed;
}

// ---------------------------------------------------------------------------
// Similarity GEMM + argmax. Block = 4 waves, owns 16 z rows; each wave sweeps
// codebook N-tiles of 16 with V_WMMA_F32_16X16X32_BF16 over K=256.
// N-tile loop is unrolled by 2 with two distinct B register buffers so the
// next tile's 16 b128 loads overlap the current tile's 8-WMMA chain.
// ---------------------------------------------------------------------------
__global__ __launch_bounds__(128, 1) void vq_argmax_kernel(
    const unsigned short* __restrict__ znb,
    const unsigned short* __restrict__ enb,
    int* __restrict__ out_idx) {
    __shared__ unsigned short zt[16 * DIM];   // 8 KB z tile (bf16)
    __shared__ float sval[4][16];
    __shared__ int   sidx[4][16];

    const int tid   = threadIdx.x;
    const int wave  = tid >> 5;
    const int lane  = tid & 31;
    const int mtile = blockIdx.x;

    // Stage z tile into LDS: 4096 ushorts = 512 uint4.
    {
        const uint4* src = (const uint4*)(znb + (size_t)mtile * 16 * DIM);
        uint4* dst = (uint4*)zt;
        for (int i = tid; i < 512; i += 128) dst[i] = src[i];
    }
    __syncthreads();

    // A fragments (16x32 bf16 each): lane L holds row M=L&15.
    // Lanes 0-15: K = kt*32 + {0..7, 16..23}; lanes 16-31: K = kt*32 + {8..15, 24..31}.
    union Frag { v16bf v; uint4 q[2]; };
    const int mrow  = lane & 15;
    const int kaoff = (lane >> 4) ? 8 : 0;
    Frag afrag[KTILES];
    #pragma unroll
    for (int kt = 0; kt < KTILES; ++kt) {
        const unsigned short* base = zt + mrow * DIM + kt * 32 + kaoff;
        afrag[kt].q[0] = *(const uint4*)(base);
        afrag[kt].q[1] = *(const uint4*)(base + 16);
    }

    // Per-lane running argmax: slot r -> row M = r + 8*(lane>=16), column
    // N = t*16 + (lane&15). Tiles are processed in increasing order, so a
    // strict > update preserves argmin's first-index tie semantics per lane.
    float bestv[8];
    int   besti[8];
    #pragma unroll
    for (int r = 0; r < 8; ++r) { bestv[r] = -3.0e38f; besti[r] = 0x7fffffff; }

    const int lane15 = lane & 15;
    const int kboff  = (lane >> 4) ? 8 : 0;   // B: lanes 0-15 K={0..7,16..23}, lanes 16-31 K={8..15,24..31}

    auto loadB = [&](Frag* b, int t) {
        const unsigned short* p = enb + ((size_t)t * 16 + lane15) * DIM + kboff;
        #pragma unroll
        for (int kt = 0; kt < KTILES; ++kt) {
            b[kt].q[0] = *(const uint4*)(p + kt * 32);
            b[kt].q[1] = *(const uint4*)(p + kt * 32 + 16);
        }
    };
    auto computeTile = [&](const Frag* b, int t) {
        v8f c = {0.f, 0.f, 0.f, 0.f, 0.f, 0.f, 0.f, 0.f};
        #pragma unroll
        for (int kt = 0; kt < KTILES; ++kt) {
            c = __builtin_amdgcn_wmma_f32_16x16x32_bf16(
                    false, afrag[kt].v, false, b[kt].v, (short)0, c, false, false);
        }
        const int ncol = t * 16 + lane15;
        #pragma unroll
        for (int r = 0; r < 8; ++r) {
            float v = c[r];
            besti[r] = (v > bestv[r]) ? ncol : besti[r];
            bestv[r] = fmaxf(bestv[r], v);
        }
    };

    // Each wave covers tiles {wave, wave+4, ..., wave+508}: 128 tiles, unroll 2.
    Frag b0[KTILES], b1[KTILES];
    loadB(b0, wave);
    for (int i = 0; i < 128; i += 2) {
        const int t0 = wave + i * 4;
        const int t1 = t0 + 4;
        const int t2 = t0 + 8;
        loadB(b1, t1);                              // t1 <= 511: always in range
        computeTile(b0, t0);
        loadB(b0, (t2 < NTILES) ? t2 : t0);         // tail: dummy in-bounds load
        computeTile(b1, t1);
    }

    // Butterfly argmax across the 16-lane half-group (lanes share the M rows).
    #pragma unroll
    for (int r = 0; r < 8; ++r) {
        float v = bestv[r]; int i = besti[r];
        #pragma unroll
        for (int m = 1; m <= 8; m <<= 1) {
            float ov = __shfl_xor(v, m, 32);
            int   oi = __shfl_xor(i, m, 32);
            if (ov > v || (ov == v && oi < i)) { v = ov; i = oi; }
        }
        bestv[r] = v; besti[r] = i;
    }
    if ((lane & 15) == 0) {
        const int mbase = (lane >> 4) * 8;
        #pragma unroll
        for (int r = 0; r < 8; ++r) {
            sval[wave][mbase + r] = bestv[r];
            sidx[wave][mbase + r] = besti[r];
        }
    }
    __syncthreads();

    if (tid < 16) {
        float v = sval[0][tid]; int i = sidx[0][tid];
        #pragma unroll
        for (int w = 1; w < 4; ++w) {
            float ov = sval[w][tid]; int oi = sidx[w][tid];
            if (ov > v || (ov == v && oi < i)) { v = ov; i = oi; }
        }
        out_idx[mtile * 16 + tid] = i;
    }
}

// ---------------------------------------------------------------------------
// Gather z_q = l2norm(codebook[idx]) in f32, write output, accumulate loss.
// One wave per z row.
// ---------------------------------------------------------------------------
__global__ __launch_bounds__(128) void vq_output_kernel(
    const float* __restrict__ z, const float* __restrict__ codebook,
    const int* __restrict__ idx, float* __restrict__ out_zq,
    float* __restrict__ out_idx_f, float* __restrict__ loss_accum) {
    const int wave = threadIdx.x >> 5;
    const int lane = threadIdx.x & 31;
    const int row  = blockIdx.x * 4 + wave;
    if (row >= NZ) return;

    const int code = idx[row];
    const float4* cr = (const float4*)(codebook + (size_t)code * DIM);
    const float4* zr = (const float4*)(z + (size_t)row * DIM);
    float4 c0 = cr[lane * 2 + 0], c1 = cr[lane * 2 + 1];
    float4 z0 = zr[lane * 2 + 0], z1 = zr[lane * 2 + 1];

    float cs = c0.x*c0.x + c0.y*c0.y + c0.z*c0.z + c0.w*c0.w
             + c1.x*c1.x + c1.y*c1.y + c1.z*c1.z + c1.w*c1.w;
    float zs = z0.x*z0.x + z0.y*z0.y + z0.z*z0.z + z0.w*z0.w
             + z1.x*z1.x + z1.y*z1.y + z1.z*z1.z + z1.w*z1.w;
    #pragma unroll
    for (int m = 16; m >= 1; m >>= 1) {
        cs += __shfl_xor(cs, m, 32);
        zs += __shfl_xor(zs, m, 32);
    }
    const float cscale = 1.0f / fmaxf(sqrtf(cs), 1e-12f);
    const float zscale = 1.0f / fmaxf(sqrtf(zs), 1e-12f);

    float4 q0, q1;
    q0.x = c0.x * cscale; q0.y = c0.y * cscale; q0.z = c0.z * cscale; q0.w = c0.w * cscale;
    q1.x = c1.x * cscale; q1.y = c1.y * cscale; q1.z = c1.z * cscale; q1.w = c1.w * cscale;

    float d, s = 0.f;
    d = q0.x - z0.x * zscale; s += d * d;
    d = q0.y - z0.y * zscale; s += d * d;
    d = q0.z - z0.z * zscale; s += d * d;
    d = q0.w - z0.w * zscale; s += d * d;
    d = q1.x - z1.x * zscale; s += d * d;
    d = q1.y - z1.y * zscale; s += d * d;
    d = q1.z - z1.z * zscale; s += d * d;
    d = q1.w - z1.w * zscale; s += d * d;
    #pragma unroll
    for (int m = 16; m >= 1; m >>= 1) s += __shfl_xor(s, m, 32);

    float4* o4 = (float4*)(out_zq + (size_t)row * DIM);
    o4[lane * 2 + 0] = q0;
    o4[lane * 2 + 1] = q1;
    if (lane == 0) {
        atomicAdd(loss_accum, s);
        out_idx_f[row] = (float)code;
    }
}

__global__ void vq_finalize_kernel(const float* __restrict__ loss_accum,
                                   float* __restrict__ out_loss) {
    // loss = (BETA + 1) * mean = 1.25 * sum / (16384*256)
    out_loss[0] = 1.25f * loss_accum[0] / 4194304.0f;
}

// ---------------------------------------------------------------------------
extern "C" void kernel_launch(void* const* d_in, const int* in_sizes, int n_in,
                              void* d_out, int out_size, void* d_ws, size_t ws_size,
                              hipStream_t stream) {
    const float* z  = (const float*)d_in[0];   // [16,1024,256] f32
    const float* cb = (const float*)d_in[1];   // [8192,256] f32
    float* out = (float*)d_out;                // zq | loss | indices

    // Workspace layout (~12.1 MB)
    unsigned short* enb = (unsigned short*)d_ws;          // 8192*256 bf16 = 4 MB
    unsigned short* znb = enb + (size_t)NK * DIM;         // 16384*256 bf16 = 8 MB
    int*   idx   = (int*)(znb + (size_t)NZ * DIM);        // 64 KB
    float* accum = (float*)(idx + NZ);                    // 4 B

    hipMemsetAsync(accum, 0, sizeof(float), stream);

    vq_normalize_bf16<<<NK / 4, 128, 0, stream>>>(cb, enb, NK);
    vq_normalize_bf16<<<NZ / 4, 128, 0, stream>>>(z,  znb, NZ);
    vq_argmax_kernel<<<NZ / 16, 128, 0, stream>>>(znb, enb, idx);
    vq_output_kernel<<<NZ / 4, 128, 0, stream>>>(z, cb, idx,
                                                 out,                    // z_q_out
                                                 out + ZQ_ELEMS + 1,    // indices
                                                 accum);
    vq_finalize_kernel<<<1, 1, 0, stream>>>(accum, out + ZQ_ELEMS);     // loss
}
